// MeshGraphDecoderConcat_28535762715034
// MI455X (gfx1250) — matrix operand order repaired and checked by reading, not compiled, not run on hardware
//
#include <hip/hip_runtime.h>
#include <hip/hip_bf16.h>

typedef __attribute__((ext_vector_type(16))) _Float16 v16h;
typedef __attribute__((ext_vector_type(8)))  _Float16 v8h;
typedef __attribute__((ext_vector_type(8)))  float    v8f;

union H16 { v16h v; v8h h[2]; _Float16 e[16]; };

#define DD 128   // feature dim
#define HH 128   // hidden dim

// ---------------------------------------------------------------------------
// zero the scatter accumulator
// ---------------------------------------------------------------------------
__global__ void mgd_zero_kernel(float4* __restrict__ p, long n4) {
    long i = (long)blockIdx.x * blockDim.x + threadIdx.x;
    if (i < n4) p[i] = make_float4(0.f, 0.f, 0.f, 0.f);
}

// ---------------------------------------------------------------------------
// Edge MLP + scatter-sum.
// Block = 256 threads = 8 waves; each wave does a 16-edge M-tile.
// LDS: W1^T f16 [128][384] + W2^T f16 [128][128] + hidden stage [8][16][128]
//      = 49152 + 16384 + 16384 halves = 163840 bytes
// ---------------------------------------------------------------------------
__global__ __launch_bounds__(256) void mgd_edge_kernel(
    const float* __restrict__ efeat, const float* __restrict__ grid,
    const float* __restrict__ mesh,
    const float* __restrict__ W1, const float* __restrict__ b1,
    const float* __restrict__ W2, const float* __restrict__ b2,
    const float* __restrict__ gamma, const float* __restrict__ beta,
    const int* __restrict__ src, const int* __restrict__ dst,
    float* __restrict__ hdest, int E)
{
    extern __shared__ _Float16 lds[];
    _Float16* w1t = lds;                   // [n=128][k=384]
    _Float16* w2t = lds + 128 * 384;       // [n=128][k=128]
    _Float16* hid = w2t + 128 * 128;       // [wave][row=16][k=128]

    const int tid = threadIdx.x;

    // stage weights into LDS, transposed, converted to f16
    for (int i = tid; i < 384 * DD; i += 256) {
        int k = i >> 7, n = i & 127;
        w1t[n * 384 + k] = (_Float16)W1[i];
    }
    for (int i = tid; i < HH * DD; i += 256) {
        int k = i >> 7, n = i & 127;
        w2t[n * HH + k] = (_Float16)W2[i];
    }
    __syncthreads();

    const int wave   = tid >> 5;
    const int lane   = tid & 31;
    const int laneLo = lane & 15;
    const int hi     = lane >> 4;     // which 16-lane half
    const long e0    = ((long)blockIdx.x * 8 + wave) * 16;

    // gather source row pointers for this lane's A row (M = laneLo)
    long eA = e0 + laneLo;
    if (eA >= E) eA = E - 1;
    const int   sIdx = src[eA];
    const int   dIdx = dst[eA];
    const float* rowp[3] = { efeat + eA * DD,
                             mesh  + (long)sIdx * DD,
                             grid  + (long)dIdx * DD };

    // ---- build A fragments: 16x384 f16 in 12 chunks of K=32 ----
    // 16-bit A 16x32 layout: lanes 0-15 hold K 0..7 & 16..23, lanes 16-31
    // hold K 8..15 & 24..31 (per-half: +hi*8)
    H16 a[12];
#pragma unroll
    for (int kc = 0; kc < 12; ++kc) {
        const float* base = rowp[kc >> 2] + (kc & 3) * 32 + hi * 8;
        const float4* b4  = (const float4*)base;
        float4 f0 = b4[0], f1 = b4[1];   // k_local 0..7
        float4 f2 = b4[4], f3 = b4[5];   // k_local 16..23
        a[kc].e[0]=(_Float16)f0.x; a[kc].e[1]=(_Float16)f0.y;
        a[kc].e[2]=(_Float16)f0.z; a[kc].e[3]=(_Float16)f0.w;
        a[kc].e[4]=(_Float16)f1.x; a[kc].e[5]=(_Float16)f1.y;
        a[kc].e[6]=(_Float16)f1.z; a[kc].e[7]=(_Float16)f1.w;
        a[kc].e[8]=(_Float16)f2.x; a[kc].e[9]=(_Float16)f2.y;
        a[kc].e[10]=(_Float16)f2.z; a[kc].e[11]=(_Float16)f2.w;
        a[kc].e[12]=(_Float16)f3.x; a[kc].e[13]=(_Float16)f3.y;
        a[kc].e[14]=(_Float16)f3.z; a[kc].e[15]=(_Float16)f3.w;
    }

    // ---- GEMM1: [16x384] x [384x128] -> acc[8] (8 N-tiles of 16) ----
    v8f acc[8];
#pragma unroll
    for (int nt = 0; nt < 8; ++nt) {
        v8f cf = {};
        const _Float16* bcol = &w1t[(nt * 16 + laneLo) * 384 + hi * 8];
#pragma unroll
        for (int kc = 0; kc < 12; ++kc) {
            H16 b;
            b.h[0] = *(const v8h*)(bcol + kc * 32);
            b.h[1] = *(const v8h*)(bcol + kc * 32 + 16);
            cf = __builtin_amdgcn_wmma_f32_16x16x32_f16(
                     false, a[kc].v, false, b.v, (short)0, cf, false, false);
        }
        acc[nt] = cf;
    }

    // ---- bias + SiLU, stage hidden to LDS as f16 ----
    _Float16* hw = hid + wave * 16 * HH;
#pragma unroll
    for (int nt = 0; nt < 8; ++nt) {
        float bb = b1[nt * 16 + laneLo];
#pragma unroll
        for (int j = 0; j < 8; ++j) {
            float x = acc[nt][j] + bb;
            float s = x / (1.0f + __expf(-x));
            int row = j + hi * 8;                 // C layout row
            hw[row * HH + nt * 16 + laneLo] = (_Float16)s;
        }
    }
    __syncthreads();

    // ---- GEMM2: [16x128] x [128x128], A rebuilt from LDS ----
    H16 a2[4];
    const _Float16* hrow = hw + laneLo * HH;
#pragma unroll
    for (int kc = 0; kc < 4; ++kc) {
        const _Float16* p = hrow + kc * 32 + hi * 8;
        a2[kc].h[0] = *(const v8h*)p;        // k_local 0..7
        a2[kc].h[1] = *(const v8h*)(p + 16); // k_local 16..23
    }
    v8f y[8];
#pragma unroll
    for (int nt = 0; nt < 8; ++nt) {
        v8f cf = {};
        const _Float16* bcol = &w2t[(nt * 16 + laneLo) * HH + hi * 8];
#pragma unroll
        for (int kc = 0; kc < 4; ++kc) {
            H16 b;
            b.h[0] = *(const v8h*)(bcol + kc * 32);
            b.h[1] = *(const v8h*)(bcol + kc * 32 + 16);
            cf = __builtin_amdgcn_wmma_f32_16x16x32_f16(
                     false, a2[kc].v, false, b.v, (short)0, cf, false, false);
        }
        float bb = b2[nt * 16 + laneLo];
#pragma unroll
        for (int j = 0; j < 8; ++j) y[nt][j] = cf[j] + bb;
    }

    // ---- LayerNorm across 128 cols per row ----
    float ps[8], pq[8];
#pragma unroll
    for (int j = 0; j < 8; ++j) { ps[j] = 0.f; pq[j] = 0.f; }
#pragma unroll
    for (int nt = 0; nt < 8; ++nt)
#pragma unroll
        for (int j = 0; j < 8; ++j) { float v = y[nt][j]; ps[j] += v; pq[j] += v * v; }
#pragma unroll
    for (int off = 1; off < 16; off <<= 1)
#pragma unroll
        for (int j = 0; j < 8; ++j) {
            ps[j] += __shfl_xor(ps[j], off, 32);
            pq[j] += __shfl_xor(pq[j], off, 32);
        }
    float mu[8], rs[8];
#pragma unroll
    for (int j = 0; j < 8; ++j) {
        mu[j] = ps[j] * (1.0f / 128.0f);
        float var = pq[j] * (1.0f / 128.0f) - mu[j] * mu[j];
        rs[j] = __frsqrt_rn(var + 1e-5f);
    }

    // ---- scale/shift + atomic scatter-sum into hdest ----
    int  drow[8];
    bool vrow[8];
#pragma unroll
    for (int j = 0; j < 8; ++j) {
        long e = e0 + j + hi * 8;
        vrow[j] = (e < E);
        drow[j] = dst[vrow[j] ? e : (E - 1)];
    }
#pragma unroll
    for (int nt = 0; nt < 8; ++nt) {
        int col = nt * 16 + laneLo;
        float gg = gamma[col], bb = beta[col];
#pragma unroll
        for (int j = 0; j < 8; ++j) {
            if (vrow[j]) {
                float o = (y[nt][j] - mu[j]) * rs[j] * gg + bb;
                atomicAdd(hdest + (long)drow[j] * DD + col, o);
            }
        }
    }
}

// ---------------------------------------------------------------------------
// Node MLP + residual.
// LDS: W1^T [128][256] + W2^T [128][128] + hidden [8][16][128] = 131072 bytes
// ---------------------------------------------------------------------------
__global__ __launch_bounds__(256) void mgd_node_kernel(
    const float* __restrict__ hdest, const float* __restrict__ grid,
    const float* __restrict__ W1, const float* __restrict__ b1,
    const float* __restrict__ W2, const float* __restrict__ b2,
    const float* __restrict__ gamma, const float* __restrict__ beta,
    float* __restrict__ out, int N)
{
    extern __shared__ _Float16 lds[];
    _Float16* w1t = lds;                   // [n=128][k=256]
    _Float16* w2t = lds + 128 * 256;
    _Float16* hid = w2t + 128 * 128;

    const int tid = threadIdx.x;
    for (int i = tid; i < 256 * DD; i += 256) {
        int k = i >> 7, n = i & 127;
        w1t[n * 256 + k] = (_Float16)W1[i];
    }
    for (int i = tid; i < HH * DD; i += 256) {
        int k = i >> 7, n = i & 127;
        w2t[n * HH + k] = (_Float16)W2[i];
    }
    __syncthreads();

    const int wave   = tid >> 5;
    const int lane   = tid & 31;
    const int laneLo = lane & 15;
    const int hi     = lane >> 4;
    const long n0    = ((long)blockIdx.x * 8 + wave) * 16;

    long rA = n0 + laneLo;
    if (rA >= N) rA = N - 1;
    const float* rowp[2] = { hdest + rA * DD, grid + rA * DD };

    // A: 16x256 in 8 chunks (first 4 from hdest, next 4 from grid)
    H16 a[8];
#pragma unroll
    for (int kc = 0; kc < 8; ++kc) {
        const float* base = rowp[kc >> 2] + (kc & 3) * 32 + hi * 8;
        const float4* b4  = (const float4*)base;
        float4 f0 = b4[0], f1 = b4[1], f2 = b4[4], f3 = b4[5];
        a[kc].e[0]=(_Float16)f0.x; a[kc].e[1]=(_Float16)f0.y;
        a[kc].e[2]=(_Float16)f0.z; a[kc].e[3]=(_Float16)f0.w;
        a[kc].e[4]=(_Float16)f1.x; a[kc].e[5]=(_Float16)f1.y;
        a[kc].e[6]=(_Float16)f1.z; a[kc].e[7]=(_Float16)f1.w;
        a[kc].e[8]=(_Float16)f2.x; a[kc].e[9]=(_Float16)f2.y;
        a[kc].e[10]=(_Float16)f2.z; a[kc].e[11]=(_Float16)f2.w;
        a[kc].e[12]=(_Float16)f3.x; a[kc].e[13]=(_Float16)f3.y;
        a[kc].e[14]=(_Float16)f3.z; a[kc].e[15]=(_Float16)f3.w;
    }

    v8f acc[8];
#pragma unroll
    for (int nt = 0; nt < 8; ++nt) {
        v8f cf = {};
        const _Float16* bcol = &w1t[(nt * 16 + laneLo) * 256 + hi * 8];
#pragma unroll
        for (int kc = 0; kc < 8; ++kc) {
            H16 b;
            b.h[0] = *(const v8h*)(bcol + kc * 32);
            b.h[1] = *(const v8h*)(bcol + kc * 32 + 16);
            cf = __builtin_amdgcn_wmma_f32_16x16x32_f16(
                     false, a[kc].v, false, b.v, (short)0, cf, false, false);
        }
        acc[nt] = cf;
    }

    _Float16* hw = hid + wave * 16 * HH;
#pragma unroll
    for (int nt = 0; nt < 8; ++nt) {
        float bb = b1[nt * 16 + laneLo];
#pragma unroll
        for (int j = 0; j < 8; ++j) {
            float x = acc[nt][j] + bb;
            float s = x / (1.0f + __expf(-x));
            int row = j + hi * 8;
            hw[row * HH + nt * 16 + laneLo] = (_Float16)s;
        }
    }
    __syncthreads();

    H16 a2[4];
    const _Float16* hrow = hw + laneLo * HH;
#pragma unroll
    for (int kc = 0; kc < 4; ++kc) {
        const _Float16* p = hrow + kc * 32 + hi * 8;
        a2[kc].h[0] = *(const v8h*)p;
        a2[kc].h[1] = *(const v8h*)(p + 16);
    }
    v8f y[8];
#pragma unroll
    for (int nt = 0; nt < 8; ++nt) {
        v8f cf = {};
        const _Float16* bcol = &w2t[(nt * 16 + laneLo) * HH + hi * 8];
#pragma unroll
        for (int kc = 0; kc < 4; ++kc) {
            H16 b;
            b.h[0] = *(const v8h*)(bcol + kc * 32);
            b.h[1] = *(const v8h*)(bcol + kc * 32 + 16);
            cf = __builtin_amdgcn_wmma_f32_16x16x32_f16(
                     false, a2[kc].v, false, b.v, (short)0, cf, false, false);
        }
        float bb = b2[nt * 16 + laneLo];
#pragma unroll
        for (int j = 0; j < 8; ++j) y[nt][j] = cf[j] + bb;
    }

    // LayerNorm
    float ps[8], pq[8];
#pragma unroll
    for (int j = 0; j < 8; ++j) { ps[j] = 0.f; pq[j] = 0.f; }
#pragma unroll
    for (int nt = 0; nt < 8; ++nt)
#pragma unroll
        for (int j = 0; j < 8; ++j) { float v = y[nt][j]; ps[j] += v; pq[j] += v * v; }
#pragma unroll
    for (int off = 1; off < 16; off <<= 1)
#pragma unroll
        for (int j = 0; j < 8; ++j) {
            ps[j] += __shfl_xor(ps[j], off, 32);
            pq[j] += __shfl_xor(pq[j], off, 32);
        }
    float mu[8], rs[8];
#pragma unroll
    for (int j = 0; j < 8; ++j) {
        mu[j] = ps[j] * (1.0f / 128.0f);
        float var = pq[j] * (1.0f / 128.0f) - mu[j] * mu[j];
        rs[j] = __frsqrt_rn(var + 1e-5f);
    }

    // scale/shift + residual + store
#pragma unroll
    for (int nt = 0; nt < 8; ++nt) {
        int col = nt * 16 + laneLo;
        float gg = gamma[col], bb = beta[col];
#pragma unroll
        for (int j = 0; j < 8; ++j) {
            long row = n0 + j + hi * 8;
            if (row < N) {
                float res = grid[row * DD + col];
                out[row * DD + col] =
                    (y[nt][j] - mu[j]) * rs[j] * gg + bb + res;
            }
        }
    }
}

// ---------------------------------------------------------------------------
extern "C" void kernel_launch(void* const* d_in, const int* in_sizes, int n_in,
                              void* d_out, int out_size, void* d_ws, size_t ws_size,
                              hipStream_t stream) {
    const float* efeat = (const float*)d_in[0];
    const float* grid  = (const float*)d_in[1];
    const float* mesh  = (const float*)d_in[2];
    const float* eW1   = (const float*)d_in[3];
    const float* eb1   = (const float*)d_in[4];
    const float* eW2   = (const float*)d_in[5];
    const float* eb2   = (const float*)d_in[6];
    const float* eg    = (const float*)d_in[7];
    const float* ebt   = (const float*)d_in[8];
    const float* nW1   = (const float*)d_in[9];
    const float* nb1   = (const float*)d_in[10];
    const float* nW2   = (const float*)d_in[11];
    const float* nb2   = (const float*)d_in[12];
    const float* ng    = (const float*)d_in[13];
    const float* nbt   = (const float*)d_in[14];
    const int*   src   = (const int*)d_in[15];
    const int*   dst   = (const int*)d_in[16];

    float* out   = (float*)d_out;
    float* hdest = (float*)d_ws;                 // [N_GRID][128] f32 scratch

    const int E = in_sizes[15];
    const int N = in_sizes[1] / DD;

    const size_t edge_lds = (size_t)(128 * 384 + 128 * 128 + 8 * 16 * 128) * 2; // 163840
    const size_t node_lds = (size_t)(128 * 256 + 128 * 128 + 8 * 16 * 128) * 2; // 131072
    hipFuncSetAttribute((const void*)mgd_edge_kernel,
                        hipFuncAttributeMaxDynamicSharedMemorySize, (int)edge_lds);
    hipFuncSetAttribute((const void*)mgd_node_kernel,
                        hipFuncAttributeMaxDynamicSharedMemorySize, (int)node_lds);

    long n4 = (long)N * DD / 4;
    mgd_zero_kernel<<<(int)((n4 + 255) / 256), 256, 0, stream>>>((float4*)hdest, n4);

    int eblocks = (E + 127) / 128;   // 8 waves x 16 edges per block
    mgd_edge_kernel<<<eblocks, 256, edge_lds, stream>>>(
        efeat, grid, mesh, eW1, eb1, eW2, eb2, eg, ebt, src, dst, hdest, E);

    int ntiles  = (N + 15) / 16;
    int nblocks = (ntiles + 7) / 8;
    mgd_node_kernel<<<nblocks, 256, node_lds, stream>>>(
        hdest, grid, nW1, nb1, nW2, nb2, ng, nbt, out, N);
}
